// TypeOneToManyAttention_61546881352177
// MI455X (gfx1250) — compile-verified, matrix operand order checked by reading, not run on hardware
//
#include <hip/hip_runtime.h>

// ---------------- problem constants ----------------
#define BATCH  8
#define EMB    256
#define NBE    (BATCH*EMB)   // 2048 independent (b,e) problems
#define KDIM   64            // keys per problem
#define HTOT   768           // model dim
#define NHEAD  12
#define HD     64            // head dim
#define LSTR   776           // padded bf16 LDS row stride (768 + 8)

// ---------------- small helpers ----------------
typedef __attribute__((ext_vector_type(16))) __bf16 v16bf;
typedef __attribute__((ext_vector_type(2)))  __bf16 v2bf;
typedef __attribute__((ext_vector_type(8)))  float  v8f;

union AFrag { uint4 q[2]; v16bf v; };

// hardware bf16 converts (v_cvt_*bf16_f32), not bit-twiddling
__device__ __forceinline__ unsigned short f2bf(float f) {
    __bf16 h = (__bf16)f;
    return __builtin_bit_cast(unsigned short, h);
}
__device__ __forceinline__ float bf2f(unsigned short b) {
    unsigned u = ((unsigned)b) << 16;
    return __builtin_bit_cast(float, u);
}
__device__ __forceinline__ unsigned pack2(float a, float b) {
    v2bf v; v[0] = (__bf16)a; v[1] = (__bf16)b;
    return __builtin_bit_cast(unsigned, v);
}

// ============================================================
// Kernel 1: convert Wq|Wk|Wv (f32, row-major [768][768]) -> bf16 in ws
// ============================================================
__global__ __launch_bounds__(256)
void cvt_w_kernel(const float* __restrict__ Wq, const float* __restrict__ Wk,
                  const float* __restrict__ Wv, unsigned short* __restrict__ dst)
{
    const int NW = HTOT * HTOT;                         // 589824
    int gid = blockIdx.x * 256 + threadIdx.x;
    int e = gid * 4;                                    // grid covers 3*NW exactly
    int m = e / NW;
    const float* s = (m == 0) ? Wq : ((m == 1) ? Wk : Wv);
    float4 f = *(const float4*)(s + (e - m * NW));
    uint2 p; p.x = pack2(f.x, f.y); p.y = pack2(f.z, f.w);
    *(uint2*)(dst + e) = p;
}

// ============================================================
// Kernel 2: Qp[2048][768] = q_in @ Wq^T + bq   (bf16 WMMA, f32 out)
// grid = (32, 12): 64x64 output tile per block; 8 waves, 2 tiles/wave
// ============================================================
__global__ __launch_bounds__(256)
void qproj_kernel(const float* __restrict__ q_in,
                  const unsigned short* __restrict__ WqB,
                  const float* __restrict__ bq, float* __restrict__ Qp)
{
    const int lane = threadIdx.x & 31;
    const int wave = threadIdx.x >> 5;
    const int l15  = lane & 15;
    const int lh   = lane >> 4;               // 0: k 0-7,16-23; 1: k 8-15,24-31 of each 32-k step
    const int tm   = wave >> 1;
    const int tn0  = (wave & 1) * 2, tn1 = tn0 + 1;

    const int arow = blockIdx.x * 64 + tm * 16 + l15;
    const int nc0  = blockIdx.y * 64 + tn0 * 16 + l15;
    const int nc1  = nc0 + 16;

    const float* ap = q_in + (size_t)arow * HTOT;
    const unsigned short* bp0 = WqB + (size_t)nc0 * HTOT;
    const unsigned short* bp1 = WqB + (size_t)nc1 * HTOT;

    v8f acc0 = {}, acc1 = {};
#pragma unroll 2
    for (int kk = 0; kk < HTOT; kk += 32) {
        int kb = kk + lh * 8;
        float4 f0 = *(const float4*)(ap + kb);
        float4 f1 = *(const float4*)(ap + kb + 4);
        float4 f2 = *(const float4*)(ap + kb + 16);
        float4 f3 = *(const float4*)(ap + kb + 20);
        AFrag a, b0, b1;
        a.q[0].x = pack2(f0.x, f0.y); a.q[0].y = pack2(f0.z, f0.w);
        a.q[0].z = pack2(f1.x, f1.y); a.q[0].w = pack2(f1.z, f1.w);
        a.q[1].x = pack2(f2.x, f2.y); a.q[1].y = pack2(f2.z, f2.w);
        a.q[1].z = pack2(f3.x, f3.y); a.q[1].w = pack2(f3.z, f3.w);
        b0.q[0] = *(const uint4*)(bp0 + kb);
        b0.q[1] = *(const uint4*)(bp0 + kb + 16);
        b1.q[0] = *(const uint4*)(bp1 + kb);
        b1.q[1] = *(const uint4*)(bp1 + kb + 16);
        acc0 = __builtin_amdgcn_wmma_f32_16x16x32_bf16(false, a.v, false, b0.v,
                                                       (short)0, acc0, false, false);
        acc1 = __builtin_amdgcn_wmma_f32_16x16x32_bf16(false, a.v, false, b1.v,
                                                       (short)0, acc1, false, false);
    }
    float bq0 = bq[nc0], bq1 = bq[nc1];
#pragma unroll
    for (int i = 0; i < 8; ++i) {
        int m = blockIdx.x * 64 + tm * 16 + lh * 8 + i;
        Qp[(size_t)m * HTOT + nc0] = acc0[i] + bq0;
        Qp[(size_t)m * HTOT + nc1] = acc1[i] + bq1;
    }
}

// ============================================================
// Kernel 3: fused K/V projection + attention, one WG per (b,e)
// LDS: sX (bf16 64x776) | sW (bf16 64x776) | sK all-heads (bf16 64x776)
//      sQ f32[768] | sV f32[64x68] | sS f32[64] | sP f32[64]  = 318,976 B
// ============================================================
__global__ __launch_bounds__(256)
void attn_main_kernel(const float* __restrict__ k_in, const float* __restrict__ v_in,
                      const float* __restrict__ tscore,
                      const float* __restrict__ bk, const float* __restrict__ bv,
                      const unsigned short* __restrict__ WkB,
                      const unsigned short* __restrict__ WvB,
                      const float* __restrict__ Qp, float* __restrict__ out)
{
    extern __shared__ __align__(16) char smem[];
    unsigned short* sX = (unsigned short*)smem;        // 64*LSTR
    unsigned short* sW = sX + 64 * LSTR;               // 64*LSTR
    unsigned short* sK = sW + 64 * LSTR;               // 64*LSTR (all heads' K)
    float* sQ = (float*)(sK + 64 * LSTR);              // 768
    float* sV = sQ + HTOT;                             // 64*68
    float* sS = sV + 64 * 68;                          // 64
    float* sP = sS + 64;                               // 64

    const int tid  = threadIdx.x;
    const int be   = blockIdx.x;
    const int lane = tid & 31;
    const int wave = tid >> 5;
    const int l15  = lane & 15;
    const int lh   = lane >> 4;
    const int tm   = wave >> 1;
    const int tn0  = (wave & 1) * 2, tn1 = tn0 + 1;
    const int ar   = tm * 16 + l15;
    const int br0  = tn0 * 16 + l15;
    const int br1  = tn1 * 16 + l15;

    // stage projected q row (f32)
    for (int i = tid; i < HTOT; i += 256) sQ[i] = Qp[(size_t)be * HTOT + i];

    // stage Xk (f32 global -> bf16 LDS)
    {
        const float* src = k_in + (size_t)be * KDIM * HTOT;
        for (int i = tid; i < KDIM * HTOT / 4; i += 256) {
            int e = i * 4, r = e / HTOT, c = e % HTOT;
            float4 f = *(const float4*)(src + r * HTOT + c);
            uint2 p; p.x = pack2(f.x, f.y); p.y = pack2(f.z, f.w);
            *(uint2*)(sX + r * LSTR + c) = p;
        }
    }
    __syncthreads();

    // ---- K projection, all heads, keep results resident in LDS as bf16 ----
    for (int h = 0; h < NHEAD; ++h) {
        const unsigned short* wsrc = WkB + (size_t)h * HD * HTOT;
        for (int i = tid; i < HD * HTOT / 8; i += 256) {
            int e = i * 8, r = e / HTOT, c = e % HTOT;
            *(uint4*)(sW + r * LSTR + c) = *(const uint4*)(wsrc + r * HTOT + c);
        }
        if (h + 1 < NHEAD)
            __builtin_prefetch(WkB + (size_t)(h + 1) * HD * HTOT, 0, 1);
        __syncthreads();

        v8f acc0 = {}, acc1 = {};
#pragma unroll 2
        for (int kk = 0; kk < HTOT; kk += 32) {
            int kb = kk + lh * 8;
            AFrag a, b0, b1;
            a.q[0]  = *(const uint4*)(sX + ar * LSTR + kb);
            a.q[1]  = *(const uint4*)(sX + ar * LSTR + kb + 16);
            b0.q[0] = *(const uint4*)(sW + br0 * LSTR + kb);
            b0.q[1] = *(const uint4*)(sW + br0 * LSTR + kb + 16);
            b1.q[0] = *(const uint4*)(sW + br1 * LSTR + kb);
            b1.q[1] = *(const uint4*)(sW + br1 * LSTR + kb + 16);
            acc0 = __builtin_amdgcn_wmma_f32_16x16x32_bf16(false, a.v, false, b0.v,
                                                           (short)0, acc0, false, false);
            acc1 = __builtin_amdgcn_wmma_f32_16x16x32_bf16(false, a.v, false, b1.v,
                                                           (short)0, acc1, false, false);
        }
        float bk0 = bk[h * HD + br0], bk1 = bk[h * HD + br1];
#pragma unroll
        for (int i = 0; i < 8; ++i) {
            int m = tm * 16 + lh * 8 + i;
            sK[m * LSTR + h * HD + br0] = f2bf(acc0[i] + bk0);
            sK[m * LSTR + h * HD + br1] = f2bf(acc1[i] + bk1);
        }
        __syncthreads();
    }

    // stage Xv over sX
    {
        const float* src = v_in + (size_t)be * KDIM * HTOT;
        for (int i = tid; i < KDIM * HTOT / 4; i += 256) {
            int e = i * 4, r = e / HTOT, c = e % HTOT;
            float4 f = *(const float4*)(src + r * HTOT + c);
            uint2 p; p.x = pack2(f.x, f.y); p.y = pack2(f.z, f.w);
            *(uint2*)(sX + r * LSTR + c) = p;
        }
    }
    __syncthreads();

    // ---- V projection + fused attention, head by head ----
    for (int h = 0; h < NHEAD; ++h) {
        const unsigned short* wsrc = WvB + (size_t)h * HD * HTOT;
        for (int i = tid; i < HD * HTOT / 8; i += 256) {
            int e = i * 8, r = e / HTOT, c = e % HTOT;
            *(uint4*)(sW + r * LSTR + c) = *(const uint4*)(wsrc + r * HTOT + c);
        }
        if (h + 1 < NHEAD)
            __builtin_prefetch(WvB + (size_t)(h + 1) * HD * HTOT, 0, 1);
        __syncthreads();

        v8f acc0 = {}, acc1 = {};
#pragma unroll 2
        for (int kk = 0; kk < HTOT; kk += 32) {
            int kb = kk + lh * 8;
            AFrag a, b0, b1;
            a.q[0]  = *(const uint4*)(sX + ar * LSTR + kb);
            a.q[1]  = *(const uint4*)(sX + ar * LSTR + kb + 16);
            b0.q[0] = *(const uint4*)(sW + br0 * LSTR + kb);
            b0.q[1] = *(const uint4*)(sW + br0 * LSTR + kb + 16);
            b1.q[0] = *(const uint4*)(sW + br1 * LSTR + kb);
            b1.q[1] = *(const uint4*)(sW + br1 * LSTR + kb + 16);
            acc0 = __builtin_amdgcn_wmma_f32_16x16x32_bf16(false, a.v, false, b0.v,
                                                           (short)0, acc0, false, false);
            acc1 = __builtin_amdgcn_wmma_f32_16x16x32_bf16(false, a.v, false, b1.v,
                                                           (short)0, acc1, false, false);
        }
        float bv0 = bv[h * HD + br0], bv1 = bv[h * HD + br1];
#pragma unroll
        for (int i = 0; i < 8; ++i) {
            int m = tm * 16 + lh * 8 + i;
            sV[m * 68 + br0] = acc0[i] + bv0;
            sV[m * 68 + br1] = acc1[i] + bv1;
        }
        __syncthreads();

        // scores[k] = (q_h . K_h[k]) / sqrt(64)
        if (tid < KDIM) {
            float s = 0.f;
#pragma unroll 8
            for (int d = 0; d < HD; ++d)
                s += sQ[h * HD + d] * bf2f(sK[tid * LSTR + h * HD + d]);
            sS[tid] = s * 0.125f;
        }
        __syncthreads();

        // softmax over 64 keys, then post-softmax type reweighting
        if (tid < KDIM) {
            float mx = -1e30f;
            for (int j = 0; j < KDIM; ++j) mx = fmaxf(mx, sS[j]);
            float den = 0.f;
            for (int j = 0; j < KDIM; ++j) den += __expf(sS[j] - mx);
            sP[tid] = (__expf(sS[tid] - mx) / den) * tscore[(size_t)be * KDIM + tid];
        }
        __syncthreads();

        // ctx[d] = sum_k p[k] * V_h[k][d]
        if (tid < HD) {
            float c = 0.f;
            for (int k2 = 0; k2 < KDIM; ++k2) c += sP[k2] * sV[k2 * 68 + tid];
            out[(size_t)be * HTOT + h * HD + tid] = c;
        }
        __syncthreads();
    }
}

// ============================================================
// launcher
// ============================================================
extern "C" void kernel_launch(void* const* d_in, const int* in_sizes, int n_in,
                              void* d_out, int out_size, void* d_ws, size_t ws_size,
                              hipStream_t stream)
{
    const float* q_in  = (const float*)d_in[0];
    const float* k_in  = (const float*)d_in[1];
    const float* v_in  = (const float*)d_in[2];
    const float* tsc   = (const float*)d_in[3];
    const float* Wq    = (const float*)d_in[4];
    const float* bq    = (const float*)d_in[5];
    const float* Wk    = (const float*)d_in[6];
    const float* bk    = (const float*)d_in[7];
    const float* Wv    = (const float*)d_in[8];
    const float* bv    = (const float*)d_in[9];
    float* out = (float*)d_out;

    const size_t NW = (size_t)HTOT * HTOT;                 // 589824 elems / matrix
    unsigned short* WB = (unsigned short*)d_ws;            // [Wq|Wk|Wv] bf16
    float* Qp = (float*)((char*)d_ws + 3 * NW * sizeof(unsigned short));

    // 1) weights f32 -> bf16 (grid covers 3*NW/4 threads exactly: 1728*256*4)
    cvt_w_kernel<<<1728, 256, 0, stream>>>(Wq, Wk, Wv, WB);

    // 2) q projection GEMM: [2048x768] = q_in @ Wq^T + bq
    qproj_kernel<<<dim3(32, 12), 256, 0, stream>>>(q_in, WB, bq, Qp);

    // 3) fused K/V projection + attention
    const unsigned smem_bytes =
        3u * 64u * LSTR * 2u +                       // sX, sW, sK (bf16)
        (HTOT + 64u * 68u + 64u + 64u) * 4u;         // sQ, sV, sS, sP (f32)
    attn_main_kernel<<<NBE, 256, smem_bytes, stream>>>(
        k_in, v_in, tsc, bk, bv, WB + NW, WB + 2 * NW, Qp, out);
}
